// TreeRNN_82875688944084
// MI455X (gfx1250) — matrix operand order repaired and checked by reading, not compiled, not run on hardware
//
#include <hip/hip_runtime.h>
#include <hip/hip_bf16.h>

// ---------------------------------------------------------------------------
// TreeRNN for MI455X (gfx1250): bf16 WMMA GEMM per tree level.
//   level GEMM:  C[M,256] = tanh(A[M,512] @ WR[256,512]^T + b)
//   final:       hidden = relu(root @ WF^T + b); out = hidden @ WO^T + b
// Weights staged per-block into LDS via the Tensor Data Mover (TDM), using
// the D# pad feature to insert a 4-DWORD bank-stagger per row.
// ---------------------------------------------------------------------------

typedef __attribute__((ext_vector_type(16))) __bf16 v16bf;
typedef __attribute__((ext_vector_type(8)))  __bf16 v8bf;
typedef __attribute__((ext_vector_type(8)))  float  v8f;
typedef __attribute__((ext_vector_type(4)))  unsigned int u32x4;
typedef __attribute__((ext_vector_type(8)))  int  i32x8;
typedef __attribute__((ext_vector_type(4)))  int  i32x4;

#define HDIM   256
#define BATCH  256
#define LEAVES 512
#define WPAD   8            // bf16 elements of per-row LDS padding (4 DWORDs)

// ----- WMMA fragment loaders (wave32, 16x16x32 bf16) ------------------------
// A 16x32 (MxK), row-major, lda elements.
// lanes 0-15: row M=lane, elems 0..7 -> K 0..7, elems 8..15 -> K 16..23
// lanes 16-31: row M=lane-16, K 8..15 and K 24..31.
__device__ __forceinline__ v16bf load_a_frag(const __bf16* __restrict__ A,
                                             int lda, int row0, int k0, int lane) {
    const int half = (lane >> 4) & 1;
    const int m    = lane & 15;
    const __bf16* p = A + (size_t)(row0 + m) * lda + k0 + half * 8;
    v16bf f;
    *(v8bf*)&f        = *(const v8bf*)p;         // b128
    *((v8bf*)&f + 1)  = *(const v8bf*)(p + 16);  // b128
    return f;
}

// B 32x16 (KxN): B[k][n] = W[h0+n][k0+k], read from the padded LDS slice.
// lanes 0-15 col N=lane cover K 0..7,16..23; lanes 16-31 cover K 8..15,24..31.
__device__ __forceinline__ v16bf load_b_lds(const __bf16* wlds, int strideE,
                                            int hrel, int k0, int lane) {
    const int n  = lane & 15;
    const int kb = (lane >> 4) * 8;
    const __bf16* p = wlds + (hrel + n) * strideE + k0 + kb;
    v16bf f;
    *(v8bf*)&f       = *(const v8bf*)p;          // ds b128
    *((v8bf*)&f + 1) = *(const v8bf*)(p + 16);   // ds b128
    return f;
}

// ----- fused GEMM + bias + activation --------------------------------------
// Block: 256 threads = 8 waves. Wave w computes 32 rows x 64 cols.
// Block tile: 256 rows x 64 cols. Grid: (M/256, 4). act: 0 = tanh, 1 = relu.
__global__ __launch_bounds__(256) void wmma_gemm_act(
    const __bf16* __restrict__ A, const __bf16* __restrict__ W,
    const float* __restrict__ bias, __bf16* __restrict__ out,
    int K, int act)
{
    __shared__ __align__(16) __bf16 wlds[64 * (2 * HDIM + WPAD)];  // 65 KiB max
    const int strideE = K + WPAD;
    const int wave = threadIdx.x >> 5;
    const int lane = threadIdx.x & 31;
    const int row0 = blockIdx.x * 256 + wave * 32;
    const int h0   = blockIdx.y * 64;

    // ---- stage 64 x K weight slice into LDS --------------------------------
#if defined(__gfx1250__) && __has_builtin(__builtin_amdgcn_tensor_load_to_lds)
    if (wave == 0) {
        const unsigned lds_base = (unsigned)(size_t)(&wlds[0]);   // addr[31:0] = LDS offset
        const unsigned long long ga =
            (unsigned long long)(size_t)(W + (size_t)h0 * K);
        const int padi = (K == 512) ? 7 : 6;      // 256 / 128 DWORDs per row
        // D# group 0: count=1 | lds_addr | global_addr | type=2
        u32x4 g0 = { 1u, lds_base, (unsigned)ga,
                     (unsigned)((ga >> 32) & 0x01FFFFFFull) | (2u << 30) };
        // D# group 1: data_size=2B, pad_enable, pad_interval=row, pad=4 DWORDs
        i32x8 g1 = { (1 << 16) | (1 << 20) | (padi << 22) | (3 << 25),
                     (K & 0xFFFF) << 16,          // tensor_dim0 lo
                     (64 << 16),                  // tensor_dim0 hi | tensor_dim1=64
                     (K << 16),                   // tensor_dim1 hi | tile_dim0=K
                     64,                          // tile_dim1=64, tile_dim2=0
                     K,                           // tensor_dim0_stride lo
                     0, 0 };
        i32x4 g2 = { 0, 0, 0, 0 };
        i32x4 g3 = { 0, 0, 0, 0 };
        i32x8 g4 = { 0, 0, 0, 0, 0, 0, 0, 0 };
        __builtin_amdgcn_tensor_load_to_lds(g0, g1, g2, g3, g4, 0);
        __builtin_amdgcn_s_wait_tensorcnt((unsigned short)0);
    }
    __syncthreads();
#else
    {
        const int chunks = K / 8;                 // 16B chunks per row
        for (int c = threadIdx.x; c < 64 * chunks; c += 256) {
            const int r = c / chunks, q = c % chunks;
            *(v8bf*)&wlds[r * strideE + q * 8] =
                *(const v8bf*)(W + (size_t)(h0 + r) * K + q * 8);
        }
        __syncthreads();
    }
#endif

    // ---- main K loop -------------------------------------------------------
    v8f acc[2][4];
    #pragma unroll
    for (int rt = 0; rt < 2; ++rt)
        #pragma unroll
        for (int t = 0; t < 4; ++t) acc[rt][t] = v8f{};

    #pragma unroll 2
    for (int k0 = 0; k0 < K; k0 += 32) {
        const v16bf a0 = load_a_frag(A, K, row0,      k0, lane);
        const v16bf a1 = load_a_frag(A, K, row0 + 16, k0, lane);
        #pragma unroll
        for (int t = 0; t < 4; ++t) {
            const v16bf b = load_b_lds(wlds, strideE, 16 * t, k0, lane);
            acc[0][t] = __builtin_amdgcn_wmma_f32_16x16x32_bf16(
                false, a0, false, b, (short)0, acc[0][t], false, false);
            acc[1][t] = __builtin_amdgcn_wmma_f32_16x16x32_bf16(
                false, a1, false, b, (short)0, acc[1][t], false, false);
        }
    }

    // ---- epilogue: bias + activation + bf16 store --------------------------
    const int n  = lane & 15;
    const int mo = (lane >> 4) * 8;
    #pragma unroll
    for (int rt = 0; rt < 2; ++rt) {
        #pragma unroll
        for (int t = 0; t < 4; ++t) {
            const int h = h0 + 16 * t + n;
            const float bv = bias[h];
            #pragma unroll
            for (int r = 0; r < 8; ++r) {
                float x = acc[rt][t][r] + bv;
                x = act ? fmaxf(x, 0.0f) : tanhf(x);
                out[(size_t)(row0 + rt * 16 + mo + r) * HDIM + h] = (__bf16)x;
            }
        }
    }
}

// ----- embed gather + f32->bf16 --------------------------------------------
__global__ __launch_bounds__(64) void gather_embed_bf16(
    const int* __restrict__ tokens, const float* __restrict__ table,
    __bf16* __restrict__ e0)
{
    const int row = blockIdx.x;              // 0 .. B*L-1
    const int t   = tokens[row];
    const float4* src = (const float4*)(table + (size_t)t * HDIM);
    __bf16* dst = e0 + (size_t)row * HDIM;
    const float4 v = src[threadIdx.x];
    __bf16 h4[4] = {(__bf16)v.x, (__bf16)v.y, (__bf16)v.z, (__bf16)v.w};
    *(uint2*)(dst + 4 * threadIdx.x) = *(uint2*)h4;
}

__global__ void f32_to_bf16_kernel(const float* __restrict__ src,
                                   __bf16* __restrict__ dst, int nElem) {
    const int i = blockIdx.x * blockDim.x + threadIdx.x;
    if (i < nElem) dst[i] = (__bf16)src[i];
}

// ----- 5-class head (tiny: 256x5 outputs, f32) -----------------------------
__global__ __launch_bounds__(256) void out_head(
    const __bf16* __restrict__ hidden, const float* __restrict__ WO,
    const float* __restrict__ WOb, float* __restrict__ out)
{
    const int idx = blockIdx.x * blockDim.x + threadIdx.x;
    if (idx >= BATCH * 5) return;
    const int b = idx / 5, c = idx % 5;
    const __bf16* h = hidden + (size_t)b * HDIM;
    const float*  w = WO + (size_t)c * HDIM;
    float s = WOb[c];
    #pragma unroll 8
    for (int d = 0; d < HDIM; ++d) s += (float)h[d] * w[d];
    out[idx] = s;
}

// ---------------------------------------------------------------------------
extern "C" void kernel_launch(void* const* d_in, const int* in_sizes, int n_in,
                              void* d_out, int out_size, void* d_ws, size_t ws_size,
                              hipStream_t stream) {
    (void)in_sizes; (void)n_in; (void)out_size; (void)ws_size;

    const int*   tokens = (const int*)  d_in[0];
    const float* embed  = (const float*)d_in[1];
    const float* WRw    = (const float*)d_in[2];  // (256, 512)
    const float* WRb    = (const float*)d_in[3];  // (256,)
    const float* WFw    = (const float*)d_in[4];  // (256, 256)
    const float* WFb    = (const float*)d_in[5];  // (256,)
    const float* WOw    = (const float*)d_in[6];  // (5, 256)
    const float* WOb    = (const float*)d_in[7];  // (5,)
    float* out = (float*)d_out;

    // workspace layout (bytes)
    char* base = (char*)d_ws;
    __bf16* buf0 = (__bf16*)(base + 0);                    // 64 MiB
    __bf16* buf1 = (__bf16*)(base + (size_t)67108864);     // 32 MiB
    __bf16* wWR  = (__bf16*)(base + (size_t)100663296);    // 256 KiB
    __bf16* wWF  = (__bf16*)(base + (size_t)100925440);    // 128 KiB
    __bf16* hid  = (__bf16*)(base + (size_t)101056512);    // 128 KiB

    // weights -> bf16 (deterministic, once per launch)
    f32_to_bf16_kernel<<<(HDIM * 2 * HDIM + 255) / 256, 256, 0, stream>>>(WRw, wWR, HDIM * 2 * HDIM);
    f32_to_bf16_kernel<<<(HDIM * HDIM + 255) / 256, 256, 0, stream>>>(WFw, wWF, HDIM * HDIM);

    // leaves: gather + convert
    gather_embed_bf16<<<BATCH * LEAVES, 64, 0, stream>>>(tokens, embed, buf0);

    // tree reduction: 9 levels, each a GEMM [M,512]x[512,256]^T with tanh
    const __bf16* cur = buf0;
    int n = LEAVES, level = 0;
    while (n > 1) {
        const int Mout = BATCH * (n >> 1);
        __bf16* dst = (level & 1) ? buf0 : buf1;
        dim3 grid(Mout / 256, 4);
        wmma_gemm_act<<<grid, 256, 0, stream>>>(cur, wWR, WRb, dst, 2 * HDIM, 0);
        cur = dst; n >>= 1; ++level;
    }

    // FF layer: hidden = relu(root @ WF^T + b), M=256, K=256
    wmma_gemm_act<<<dim3(1, 4), 256, 0, stream>>>(cur, wWF, WFb, hid, HDIM, 1);

    // head: out = hidden @ WO^T + b  (256 x 5, f32)
    out_head<<<(BATCH * 5 + 255) / 256, 256, 0, stream>>>(hid, WOw, WOb, out);
}
